// TensorODEBLOCK_11570641895919
// MI455X (gfx1250) — matrix operand order repaired and checked by reading, not compiled, not run on hardware
//
#include <hip/hip_runtime.h>
#include <hip/hip_bf16.h>
#include <math.h>

// ---------------------------------------------------------------------------
// TensorODE block on MI455X (gfx1250, wave32, WMMA, async-to-LDS)
//   A0 = x @ P                         (f32 WMMA 16x16x4)
//   RK45 (Dormand-Prince, 20 masked steps, grid-wide accept/reject)
//     f(t,A)[b,i] = sum_j sin(A+t)*C[i,j,0] + cos(A+t)*C[i,j,1]
//                 = (G [512x1024] @ W^T) / 1024   with W = 1024*C as f16,
//                   W row-major [i][kk] == column-major B  (f16 WMMA 16x16x32)
//   out = Af @ F                       (f32 WMMA 16x16x4)
// ---------------------------------------------------------------------------

typedef __attribute__((ext_vector_type(16))) _Float16 v16h;
typedef __attribute__((ext_vector_type(8)))  _Float16 v8h;
typedef __attribute__((ext_vector_type(8)))  float    v8f;
typedef __attribute__((ext_vector_type(2)))  float    v2f;

#define BB    512
#define DT    512
#define K2    1024           // 2*DT (sin/cos interleaved K)
#define NEL   (BB*DT)

// scalar state slots (device)
enum { S_T=0, S_H, S_DONE, S_HS, S_COMMIT, S_ERR, S_D0, S_D1, S_D2, S_H0, S_NSLOTS };

// ---------------------------------------------------------------------------
__global__ void k_zero_scalars(float* S) {
  if (threadIdx.x < S_NSLOTS) S[threadIdx.x] = 0.0f;
}

// W = 1024 * C as f16, same flat layout: W[i*1024 + 2j+s] = 1024*C[i,j,s].
// Read as [i][kk], this IS the (column-major) B matrix of the f-eval GEMM.
__global__ void k_convert_C(const float* __restrict__ C, _Float16* __restrict__ W) {
  int idx = blockIdx.x * blockDim.x + threadIdx.x;
  if (idx >= DT * DT * 2) return;
  W[idx] = (_Float16)(C[idx] * 1024.0f);
}

// ---------------------------------------------------------------------------
// f32 GEMM: out[M,N] = A[M,K] @ B[K,N], one wave per 16x16 tile, K step 4.
// v_wmma_f32_16x16x4_f32: A frag = 2 f32/lane (lanes<16: K=k..k+1; lanes>=16: K=k+2..k+3)
__global__ __launch_bounds__(128)
void k_gemm_f32(const float* __restrict__ A, const float* __restrict__ Bm,
                float* __restrict__ out, int M, int N, int K) {
  int lane = threadIdx.x & 31, wave = threadIdx.x >> 5;
  int tile = blockIdx.x * 4 + wave;
  int ntile = N >> 4;
  int tm = tile / ntile, tn = tile % ntile;
  if (tm >= (M >> 4)) return;                 // uniform per wave
  int half = lane >> 4, r = lane & 15;
  int row = tm * 16 + r, col = tn * 16 + r;
  v8f acc = {};
  for (int k = 0; k < K; k += 4) {
    int ka = k + half * 2;
    v2f a, b;
    a.x = A[row * K + ka];
    a.y = A[row * K + ka + 1];
    b.x = Bm[ka * N + col];
    b.y = Bm[(ka + 1) * N + col];
    acc = __builtin_amdgcn_wmma_f32_16x16x4_f32(false, a, false, b, (short)0, acc,
                                                false, false);
  }
  for (int v = 0; v < 8; ++v)
    out[(tm * 16 + half * 8 + v) * N + col] = acc[v];
}

// ---------------------------------------------------------------------------
// f16 GEMM: out[512,512] = alpha * (G[512,1024] @ B),  B[kk,i] = W[i*1024+kk]
// 64x64 tile/block, 128 threads (4 waves, 2x2 of 32x32 subtiles).
// Double-buffered LDS staging via GLOBAL_LOAD_ASYNC_TO_LDS_B128 (ASYNCcnt).
__global__ __launch_bounds__(128)
void k_gemm_f16(const _Float16* __restrict__ G, const _Float16* __restrict__ W,
                float* __restrict__ out, float alpha) {
  __shared__ _Float16 As[2][64 * 32];    // [buf][row 64][k 32]
  __shared__ _Float16 Bst[2][64 * 40];   // [buf][col 64][k 32 + pad 8] (80B rows, 16B aligned)
  const int tid = threadIdx.x, lane = tid & 31, wave = tid >> 5;
  const int m0 = blockIdx.y * 64, n0 = blockIdx.x * 64;
  const int half = lane >> 4, r = lane & 15;
  const int mw = (wave >> 1) * 32, nw = (wave & 1) * 32;

  v8f acc00 = {}, acc01 = {}, acc10 = {}, acc11 = {};

  // Issue async global->LDS copies for one 64x32 A tile + 64x32 B tile.
  // 2 x 16B chunks per tile per thread -> 4 async instructions per wave per buffer.
  auto issue = [&](int buf, int kc) {
    for (int c = tid; c < 256; c += 128) {
      int row = c >> 2, cc = (c & 3) * 8;                    // cc in f16 elements
      unsigned aoff = (unsigned)(uintptr_t)&As[buf][row * 32 + cc];
      const _Float16* ga = &G[(m0 + row) * K2 + kc + cc];
      asm volatile("global_load_async_to_lds_b128 %0, %1, off"
                   :: "v"(aoff), "v"(ga) : "memory");
      unsigned boff = (unsigned)(uintptr_t)&Bst[buf][row * 40 + cc];
      const _Float16* gb = &W[(n0 + row) * K2 + kc + cc];    // row = output col
      asm volatile("global_load_async_to_lds_b128 %0, %1, off"
                   :: "v"(boff), "v"(gb) : "memory");
    }
  };

  issue(0, 0);
  int cur = 0;
  for (int kc = 0; kc < K2; kc += 32, cur ^= 1) {
    if (kc) __syncthreads();             // all waves done computing the buffer we refill
    if (kc + 32 < K2) {
      issue(cur ^ 1, kc + 32);
      asm volatile("s_wait_asynccnt 0x4" ::: "memory");  // retire current buffer's 4 copies
    } else {
      asm volatile("s_wait_asynccnt 0x0" ::: "memory");
    }
    __syncthreads();                     // current buffer visible to all waves

    const _Float16* as = As[cur];
    const _Float16* bs = Bst[cur];
    // A fragments (ISA 16-bit A 16x32): half 0 -> K {0..7,16..23}, half 1 -> K {8..15,24..31}
    v8h l0 = *(const v8h*)&as[(mw + r) * 32 + half * 8];
    v8h h0 = *(const v8h*)&as[(mw + r) * 32 + 16 + half * 8];
    v16h a0 = __builtin_shufflevector(l0, h0, 0,1,2,3,4,5,6,7,8,9,10,11,12,13,14,15);
    v8h l1 = *(const v8h*)&as[(mw + 16 + r) * 32 + half * 8];
    v8h h1 = *(const v8h*)&as[(mw + 16 + r) * 32 + 16 + half * 8];
    v16h a1 = __builtin_shufflevector(l1, h1, 0,1,2,3,4,5,6,7,8,9,10,11,12,13,14,15);
    // B fragments: half 0 -> K {0..7,16..23}, half 1 -> K {8..15,24..31} (same as A)
    v8h bl0 = *(const v8h*)&bs[(nw + r) * 40 + half * 8];
    v8h bh0 = *(const v8h*)&bs[(nw + r) * 40 + 16 + half * 8];
    v16h b0 = __builtin_shufflevector(bl0, bh0, 0,1,2,3,4,5,6,7,8,9,10,11,12,13,14,15);
    v8h bl1 = *(const v8h*)&bs[(nw + 16 + r) * 40 + half * 8];
    v8h bh1 = *(const v8h*)&bs[(nw + 16 + r) * 40 + 16 + half * 8];
    v16h b1 = __builtin_shufflevector(bl1, bh1, 0,1,2,3,4,5,6,7,8,9,10,11,12,13,14,15);

    acc00 = __builtin_amdgcn_wmma_f32_16x16x32_f16(false, a0, false, b0, (short)0, acc00, false, false);
    acc01 = __builtin_amdgcn_wmma_f32_16x16x32_f16(false, a0, false, b1, (short)0, acc01, false, false);
    acc10 = __builtin_amdgcn_wmma_f32_16x16x32_f16(false, a1, false, b0, (short)0, acc10, false, false);
    acc11 = __builtin_amdgcn_wmma_f32_16x16x32_f16(false, a1, false, b1, (short)0, acc11, false, false);
  }

  for (int v = 0; v < 8; ++v) {
    int mr = m0 + mw + half * 8 + v;
    out[mr * DT + n0 + nw + r]             = acc00[v] * alpha;
    out[mr * DT + n0 + nw + 16 + r]        = acc01[v] * alpha;
    out[(mr + 16) * DT + n0 + nw + r]      = acc10[v] * alpha;
    out[(mr + 16) * DT + n0 + nw + 16 + r] = acc11[v] * alpha;
  }
}

// ---------------------------------------------------------------------------
// G[b,2j]=sin(phase), G[b,2j+1]=cos(phase);  phase = y + hmul*(sum ci*ki) + tbase + c_t*hmul
__global__ void k_prepare(const float* __restrict__ y,
                          const float* k1, const float* k2, const float* k3,
                          const float* k4, const float* k5, const float* k6,
                          float c1, float c2, float c3, float c4, float c5, float c6,
                          const float* __restrict__ S, int idx_h, int idx_t, float c_t,
                          _Float16* __restrict__ G) {
  int i = blockIdx.x * blockDim.x + threadIdx.x;
  if (i >= NEL) return;
  float hm = (idx_h >= 0) ? S[idx_h] : 0.0f;
  float tb = (idx_t >= 0) ? S[idx_t] : 0.0f;
  float a = 0.0f;
  if (c1 != 0.0f) a += c1 * k1[i];
  if (c2 != 0.0f) a += c2 * k2[i];
  if (c3 != 0.0f) a += c3 * k3[i];
  if (c4 != 0.0f) a += c4 * k4[i];
  if (c5 != 0.0f) a += c5 * k5[i];
  if (c6 != 0.0f) a += c6 * k6[i];
  float phase = y[i] + hm * a + tb + c_t * hm;
  float sv, cv;
  sincosf(phase, &sv, &cv);
  G[2 * i]     = (_Float16)sv;
  G[2 * i + 1] = (_Float16)cv;
}

// y_new = y + hs*(b1 k1 + b3 k3 + b4 k4 + b5 k5 + b6 k6); G7 = sincos(y_new + t + hs); zero errsum
__global__ void k_finalize_y(const float* __restrict__ y, const float* __restrict__ k1,
                             const float* __restrict__ k3, const float* __restrict__ k4,
                             const float* __restrict__ k5, const float* __restrict__ k6,
                             float* __restrict__ S, float* __restrict__ ynew,
                             _Float16* __restrict__ G) {
  int i = blockIdx.x * blockDim.x + threadIdx.x;
  if (blockIdx.x == 0 && threadIdx.x == 0) S[S_ERR] = 0.0f;
  if (i >= NEL) return;
  float hs = S[S_HS], t = S[S_T];
  float yn = y[i] + hs * ((35.0f/384.0f)*k1[i] + (500.0f/1113.0f)*k3[i] +
                          (125.0f/192.0f)*k4[i] - (2187.0f/6784.0f)*k5[i] +
                          (11.0f/84.0f)*k6[i]);
  ynew[i] = yn;
  float sv, cv;
  sincosf(yn + t + hs, &sv, &cv);
  G[2 * i]     = (_Float16)sv;
  G[2 * i + 1] = (_Float16)cv;
}

// ---------------------------------------------------------------------------
// reductions (256 blocks x 256 threads, grid-stride, LDS tree + global atomic)
__device__ __forceinline__ void block_reduce_add(float v, float* slot) {
  __shared__ float red[256];
  int tid = threadIdx.x;
  red[tid] = v;
  __syncthreads();
  for (int s = 128; s > 0; s >>= 1) {
    if (tid < s) red[tid] += red[tid + s];
    __syncthreads();
  }
  if (tid == 0) atomicAdd(slot, red[0]);
}

__global__ void k_reduce_init(const float* __restrict__ y0, const float* __restrict__ f0,
                              float* __restrict__ S) {
  float s0 = 0.0f, s1 = 0.0f;
  for (int i = blockIdx.x * blockDim.x + threadIdx.x; i < NEL; i += gridDim.x * blockDim.x) {
    float yv = y0[i], fv = f0[i];
    float sc = 1e-6f + 1e-3f * fabsf(yv);
    float a = yv / sc, b = fv / sc;
    s0 += a * a; s1 += b * b;
  }
  block_reduce_add(s0, &S[S_D0]);
  __syncthreads();
  block_reduce_add(s1, &S[S_D1]);
}

__global__ void k_reduce_d2(const float* __restrict__ y0, const float* __restrict__ f0,
                            const float* __restrict__ f1, float* __restrict__ S) {
  float s = 0.0f;
  for (int i = blockIdx.x * blockDim.x + threadIdx.x; i < NEL; i += gridDim.x * blockDim.x) {
    float sc = 1e-6f + 1e-3f * fabsf(y0[i]);
    float d = (f1[i] - f0[i]) / sc;
    s += d * d;
  }
  block_reduce_add(s, &S[S_D2]);
}

__global__ void k_reduce_err(const float* __restrict__ k1, const float* __restrict__ k3,
                             const float* __restrict__ k4, const float* __restrict__ k5,
                             const float* __restrict__ k6, const float* __restrict__ k7,
                             const float* __restrict__ y, const float* __restrict__ yn,
                             float* __restrict__ S) {
  float hs = S[S_HS];
  float s = 0.0f;
  for (int i = blockIdx.x * blockDim.x + threadIdx.x; i < NEL; i += gridDim.x * blockDim.x) {
    float err = hs * ((71.0f/57600.0f)*k1[i] - (71.0f/16695.0f)*k3[i] +
                      (71.0f/1920.0f)*k4[i] - (17253.0f/339200.0f)*k5[i] +
                      (22.0f/525.0f)*k6[i] - (1.0f/40.0f)*k7[i]);
    float sc = 1e-6f + 1e-3f * fmaxf(fabsf(y[i]), fabsf(yn[i]));
    float r = err / sc;
    s += r * r;
  }
  block_reduce_add(s, &S[S_ERR]);
}

// ---------------------------------------------------------------------------
// scalar control (1 thread)
__global__ void k_init_control1(float* S) {
  float d0 = sqrtf(S[S_D0] / (float)NEL);
  float d1 = sqrtf(S[S_D1] / (float)NEL);
  S[S_H0] = (d0 < 1e-5f || d1 < 1e-5f) ? 1e-6f : 0.01f * d0 / d1;
  S[S_D2] = 0.0f;
}

__global__ void k_init_control2(float* S) {
  float h0 = S[S_H0];
  float d1 = sqrtf(S[S_D1] / (float)NEL);
  float d2 = sqrtf(S[S_D2] / (float)NEL) / h0;
  float dmax = fmaxf(d1, d2);
  float h1 = (dmax <= 1e-15f) ? fmaxf(1e-6f, h0 * 1e-3f) : powf(0.01f / dmax, 0.2f);
  float h = fminf(fminf(100.0f * h0, h1), 1.0f);
  S[S_T] = 0.0f;
  S[S_H] = h;
  S[S_DONE] = 0.0f;
  S[S_HS] = fminf(h, 1.0f);
  S[S_ERR] = 0.0f;
}

__global__ void k_step_control(float* S) {
  float en = sqrtf(S[S_ERR] / (float)NEL);
  bool accept = en < 1.0f;
  float safe = fmaxf(en, 1e-10f);
  float f = 0.9f * powf(safe, -0.2f);
  float fac = accept ? fminf(10.0f, f) : fmaxf(0.2f, f);
  float hs = S[S_HS], t = S[S_T], done = S[S_DONE];
  float t_n = accept ? (t + hs) : t;
  float h_n = hs * fac;
  bool reached = t_n >= 1.0f - 1e-7f;
  if (done == 0.0f) { S[S_T] = t_n; S[S_H] = h_n; }
  S[S_DONE] = (done != 0.0f || reached) ? 1.0f : 0.0f;
  S[S_COMMIT] = (accept && done == 0.0f) ? 1.0f : 0.0f;
  S[S_HS] = fminf(S[S_H], 1.0f - S[S_T]);
}

__global__ void k_commit(const float* __restrict__ S, float* __restrict__ y,
                         const float* __restrict__ ynew, float* __restrict__ fy,
                         const float* __restrict__ k7) {
  if (S[S_COMMIT] == 0.0f) return;
  int i = blockIdx.x * blockDim.x + threadIdx.x;
  if (i >= NEL) return;
  y[i] = ynew[i];
  fy[i] = k7[i];
}

// ---------------------------------------------------------------------------
extern "C" void kernel_launch(void* const* d_in, const int* in_sizes, int n_in,
                              void* d_out, int out_size, void* d_ws, size_t ws_size,
                              hipStream_t stream) {
  (void)in_sizes; (void)n_in; (void)out_size; (void)ws_size;
  const float* x = (const float*)d_in[0];
  const float* P = (const float*)d_in[1];
  const float* C = (const float*)d_in[2];
  const float* F = (const float*)d_in[3];
  float* out = (float*)d_out;

  char* ws = (char*)d_ws;
  size_t off = 0;
  auto alloc = [&](size_t bytes) -> void* {
    void* p = ws + off;
    off = (off + bytes + 255) & ~(size_t)255;
    return p;
  };
  float*    S   = (float*)alloc(S_NSLOTS * sizeof(float));
  float*    Y   = (float*)alloc(NEL * 4);
  float*    YN  = (float*)alloc(NEL * 4);
  float*    K1  = (float*)alloc(NEL * 4);
  float*    K2v = (float*)alloc(NEL * 4);
  float*    K3  = (float*)alloc(NEL * 4);
  float*    K4  = (float*)alloc(NEL * 4);
  float*    K5  = (float*)alloc(NEL * 4);
  float*    K6  = (float*)alloc(NEL * 4);
  float*    K7  = (float*)alloc(NEL * 4);
  _Float16* G   = (_Float16*)alloc(NEL * 2 * sizeof(_Float16));
  _Float16* W   = (_Float16*)alloc((size_t)DT * K2 * sizeof(_Float16));

  const dim3 gEl(NEL / 256), bEl(256);
  const dim3 gGemm(8, 8), bGemm(128);
  const dim3 gRed(256), bRed(256);
  const float alpha = 1.0f / 1024.0f;

  // one-time setup (re-done every call for determinism)
  k_convert_C<<<(DT * DT * 2 + 255) / 256, 256, 0, stream>>>(C, W);
  k_zero_scalars<<<1, 32, 0, stream>>>(S);

  // A0 = x @ P
  k_gemm_f32<<<256, 128, 0, stream>>>(x, P, Y, BB, DT, DT);

  // f0 = f(0, A0) -> K1
  k_prepare<<<gEl, bEl, 0, stream>>>(Y, nullptr, nullptr, nullptr, nullptr, nullptr, nullptr,
                                     0, 0, 0, 0, 0, 0, S, -1, -1, 0.0f, G);
  k_gemm_f16<<<gGemm, bGemm, 0, stream>>>(G, W, K1, alpha);

  // initial step size selection
  k_reduce_init<<<gRed, bRed, 0, stream>>>(Y, K1, S);
  k_init_control1<<<1, 1, 0, stream>>>(S);
  // f1 = f(h0, y0 + h0*f0) -> K2v
  k_prepare<<<gEl, bEl, 0, stream>>>(Y, K1, nullptr, nullptr, nullptr, nullptr, nullptr,
                                     1.0f, 0, 0, 0, 0, 0, S, S_H0, -1, 1.0f, G);
  k_gemm_f16<<<gGemm, bGemm, 0, stream>>>(G, W, K2v, alpha);
  k_reduce_d2<<<gRed, bRed, 0, stream>>>(Y, K1, K2v, S);
  k_init_control2<<<1, 1, 0, stream>>>(S);

  // 20 masked RK45 steps
  for (int step = 0; step < 20; ++step) {
    // k2
    k_prepare<<<gEl, bEl, 0, stream>>>(Y, K1, nullptr, nullptr, nullptr, nullptr, nullptr,
                                       1.0f/5.0f, 0, 0, 0, 0, 0, S, S_HS, S_T, 1.0f/5.0f, G);
    k_gemm_f16<<<gGemm, bGemm, 0, stream>>>(G, W, K2v, alpha);
    // k3
    k_prepare<<<gEl, bEl, 0, stream>>>(Y, K1, K2v, nullptr, nullptr, nullptr, nullptr,
                                       3.0f/40.0f, 9.0f/40.0f, 0, 0, 0, 0, S, S_HS, S_T, 3.0f/10.0f, G);
    k_gemm_f16<<<gGemm, bGemm, 0, stream>>>(G, W, K3, alpha);
    // k4
    k_prepare<<<gEl, bEl, 0, stream>>>(Y, K1, K2v, K3, nullptr, nullptr, nullptr,
                                       44.0f/45.0f, -56.0f/15.0f, 32.0f/9.0f, 0, 0, 0,
                                       S, S_HS, S_T, 4.0f/5.0f, G);
    k_gemm_f16<<<gGemm, bGemm, 0, stream>>>(G, W, K4, alpha);
    // k5
    k_prepare<<<gEl, bEl, 0, stream>>>(Y, K1, K2v, K3, K4, nullptr, nullptr,
                                       19372.0f/6561.0f, -25360.0f/2187.0f,
                                       64448.0f/6561.0f, -212.0f/729.0f, 0, 0,
                                       S, S_HS, S_T, 8.0f/9.0f, G);
    k_gemm_f16<<<gGemm, bGemm, 0, stream>>>(G, W, K5, alpha);
    // k6
    k_prepare<<<gEl, bEl, 0, stream>>>(Y, K1, K2v, K3, K4, K5, nullptr,
                                       9017.0f/3168.0f, -355.0f/33.0f, 46732.0f/5247.0f,
                                       49.0f/176.0f, -5103.0f/18656.0f, 0,
                                       S, S_HS, S_T, 1.0f, G);
    k_gemm_f16<<<gGemm, bGemm, 0, stream>>>(G, W, K6, alpha);
    // y_new + k7 (FSAL)
    k_finalize_y<<<gEl, bEl, 0, stream>>>(Y, K1, K3, K4, K5, K6, S, YN, G);
    k_gemm_f16<<<gGemm, bGemm, 0, stream>>>(G, W, K7, alpha);
    // error norm + scalar control + masked commit
    k_reduce_err<<<gRed, bRed, 0, stream>>>(K1, K3, K4, K5, K6, K7, Y, YN, S);
    k_step_control<<<1, 1, 0, stream>>>(S);
    k_commit<<<gEl, bEl, 0, stream>>>(S, Y, YN, K1, K7);
  }

  // out = Af @ F
  k_gemm_f32<<<256, 128, 0, stream>>>(Y, F, out, BB, DT, DT);
}
